// Dyn2DRadEncObj_74131135529183
// MI455X (gfx1250) — matrix operand order repaired and checked by reading, not compiled, not run on hardware
//
#include <hip/hip_runtime.h>
#include <hip/hip_bf16.h>
#include <stdint.h>

// ---------------------------------------------------------------------------
// Dyn2DRadEncObj  E^H D E  (frame-wise SENSE NDFT normal operator), gfx1250.
// Complex GEMMs via V_WMMA_F32_16X16X32_F16 (f16 in / f32 accumulate).
// ---------------------------------------------------------------------------

#define NXc   128
#define NYc   128
#define NTc   16
#define NCc   8
#define NRADc 2048
#define KCH   128          // k-points per chunk
#define NCHUNK (NRADc / KCH)

typedef _Float16 half16 __attribute__((ext_vector_type(16)));
typedef float    float8 __attribute__((ext_vector_type(8)));

// ---- fragment helpers -----------------------------------------------------

union FragU { half16 h; uint4 q[2]; uint32_t u[8]; };

// A-fragment (16x32 f16): row-major [M][K] source, row = M (lane%16 already
// folded in by caller), col = K0 + (lane/16)*8 ; second 16B run at +16 halves.
__device__ __forceinline__ half16 load_a_frag(const _Float16* __restrict__ base,
                                              int ld, int row, int col) {
  const _Float16* p = base + (size_t)row * ld + col;
  FragU f;
  f.q[0] = *(const uint4*)(p);
  f.q[1] = *(const uint4*)(p + 16);
  return f.h;
}

// B-fragment (32x16 f16): [N][K]-major source (K contiguous), lane holds
// column N = lane%16, K = K0 + (lane/16)*8 ; second 16B run at +16 halves.
__device__ __forceinline__ half16 load_b_frag(const _Float16* p) {
  FragU f;
  f.q[0] = *(const uint4*)(p);
  f.q[1] = *(const uint4*)(p + 16);
  return f.h;
}

__device__ __forceinline__ half16 hneg16(half16 a) {
  FragU f; f.h = a;
#pragma unroll
  for (int i = 0; i < 8; ++i) f.u[i] ^= 0x80008000u;
  return f.h;
}

__device__ __forceinline__ float8 wmma_f16(half16 a, half16 b, float8 c) {
  return __builtin_amdgcn_wmma_f32_16x16x32_f16(false, a, false, b,
                                                (short)0, c, false, false);
}

// ---- kernel 1: phase tables ----------------------------------------------
// ExR/ExI : [t][k][x] f16 planar (forward A-fragments)
// ExTR/ExTI: [t][x][k] f16 planar (adjoint A-fragments)
// EyP     : [t][k][y] packed {cos, -sin} as 2 x f16 in a u32
__global__ void build_tables(const float* __restrict__ KT,
                             _Float16* __restrict__ ExR, _Float16* __restrict__ ExI,
                             _Float16* __restrict__ ExTR, _Float16* __restrict__ ExTI,
                             uint32_t* __restrict__ EyP) {
  int idx = blockIdx.x * 256 + threadIdx.x;
  const int total = NTc * NRADc * NXc;
  if (idx >= total) return;
  int j = idx & (NXc - 1);
  int k = (idx >> 7) & (NRADc - 1);
  int t = idx >> 18;                     // / (128*2048)
  float kx = KT[(size_t)k * NTc + t];
  float ky = KT[(size_t)NRADc * NTc + (size_t)k * NTc + t];
  float coord = (float)(j - NXc / 2);
  float s, c;
  __sincosf(kx * coord, &s, &c);
  _Float16 exr = (_Float16)c, exi = (_Float16)(-s);
  ExR[idx] = exr;
  ExI[idx] = exi;
  size_t tr = ((size_t)t * NXc + j) * NRADc + k;
  ExTR[tr] = exr;
  ExTI[tr] = exi;
  __sincosf(ky * coord, &s, &c);
  union { uint32_t u; _Float16 h[2]; } p;
  p.h[0] = (_Float16)c;
  p.h[1] = (_Float16)(-s);
  EyP[idx] = p.u;
}

__global__ void zero_out(float* __restrict__ out, int n) {
  int i = blockIdx.x * 256 + threadIdx.x;
  if (i < n) out[i] = 0.0f;
}

// ---- kernel 2: fused E^H D E per (coil, frame) ----------------------------
__global__ __launch_bounds__(256)
void adaga_kernel(const float* __restrict__ X,      // (2,NX,NY,NT)
                  const float* __restrict__ CS,     // (NC,2,NX,NY)
                  const float* __restrict__ DC,     // (NRAD,NT)
                  const _Float16* __restrict__ ExR,
                  const _Float16* __restrict__ ExI,
                  const _Float16* __restrict__ ExTR,
                  const _Float16* __restrict__ ExTI,
                  const uint32_t* __restrict__ EyP,
                  float* __restrict__ out) {       // (2,NX,NY,NT)
  const int c = blockIdx.x;   // coil
  const int t = blockIdx.y;   // frame

  extern __shared__ char smem[];
  _Float16* srcTr = (_Float16*)smem;            // [y][x]  128x128
  _Float16* srcTi = srcTr + NXc * NYc;
  _Float16* uTr   = srcTi + NXc * NYc;          // [y][k]  128x128 (chunk)
  _Float16* uTi   = uTr + NXc * NYc;
  float2*   gSh   = (float2*)(uTi + NXc * NYc); // [128] dcomp-weighted k-data

  const int tid  = threadIdx.x;
  const int lane = tid & 31;
  const int w    = tid >> 5;        // wave id 0..7
  const int lo   = lane & 15;
  const int hi   = lane >> 4;

  // --- build src^T = (smap * img)^T in LDS (B operand of forward GEMM) ---
  const size_t nxyt = (size_t)NXc * NYc * NTc;
#pragma unroll 4
  for (int i = 0; i < 64; ++i) {
    int e  = tid + i * 256;               // 0..16383
    int xx = e & (NXc - 1);
    int y  = e >> 7;
    float imr = X[(size_t)xx * NYc * NTc + (size_t)y * NTc + t];
    float imi = X[nxyt + (size_t)xx * NYc * NTc + (size_t)y * NTc + t];
    float smr = CS[((size_t)c * 2 + 0) * NXc * NYc + (size_t)xx * NYc + y];
    float smi = CS[((size_t)c * 2 + 1) * NXc * NYc + (size_t)xx * NYc + y];
    srcTr[y * NXc + xx] = (_Float16)(smr * imr - smi * imi);
    srcTi[y * NXc + xx] = (_Float16)(smr * imi + smi * imr);
  }
  __syncthreads();

  const _Float16* ExRt  = ExR  + (size_t)t * NRADc * NXc;
  const _Float16* ExIt  = ExI  + (size_t)t * NRADc * NXc;
  const _Float16* ExTRt = ExTR + (size_t)t * NXc * NRADc;
  const _Float16* ExTIt = ExTI + (size_t)t * NXc * NRADc;
  const uint32_t* EyPt  = EyP  + (size_t)t * NRADc * NYc;

  // adjoint accumulators: this wave owns x rows [w*16, w*16+16), all 128 y
  float8 xAr[8], xAi[8];
#pragma unroll
  for (int n = 0; n < 8; ++n) {
#pragma unroll
    for (int r = 0; r < 8; ++r) { xAr[n][r] = 0.0f; xAi[n][r] = 0.0f; }
  }

  for (int ch = 0; ch < NCHUNK; ++ch) {
    const int kg0 = ch * KCH;

    // ---- forward GEMM: T[k,y] = Ex[k,:] @ src[:,y], chunk of 128 k-rows ---
    float8 tR[8], tI[8];
#pragma unroll
    for (int n = 0; n < 8; ++n) {
#pragma unroll
      for (int r = 0; r < 8; ++r) { tR[n][r] = 0.0f; tI[n][r] = 0.0f; }
    }
#pragma unroll
    for (int x0 = 0; x0 < NXc; x0 += 32) {
      half16 Ar  = load_a_frag(ExRt, NXc, kg0 + w * 16 + lo, x0 + hi * 8);
      half16 Ai  = load_a_frag(ExIt, NXc, kg0 + w * 16 + lo, x0 + hi * 8);
      half16 AiN = hneg16(Ai);
#pragma unroll
      for (int n = 0; n < 8; ++n) {
        half16 Br = load_b_frag(srcTr + (n * 16 + lo) * NXc + x0 + hi * 8);
        half16 Bi = load_b_frag(srcTi + (n * 16 + lo) * NXc + x0 + hi * 8);
        tR[n] = wmma_f16(Ar,  Br, tR[n]);
        tR[n] = wmma_f16(AiN, Bi, tR[n]);   // - Ai*Bi
        tI[n] = wmma_f16(Ar,  Bi, tI[n]);
        tI[n] = wmma_f16(Ai,  Br, tI[n]);
      }
    }

    // ---- kdat[k] = sum_y Ey[k,y]*T[k,y];  g = kdat * dcomp / (Nx*Ny) ------
#pragma unroll
    for (int r = 0; r < 8; ++r) {
      float sr = 0.0f, si = 0.0f;
      const int kg = kg0 + w * 16 + r + 8 * hi;     // this lane's k row
#pragma unroll
      for (int n = 0; n < 8; ++n) {
        union { uint32_t u; _Float16 h[2]; } e;
        e.u = EyPt[(size_t)kg * NYc + n * 16 + lo];
        float eyr = (float)e.h[0], eyi = (float)e.h[1];
        float tr = tR[n][r], ti = tI[n][r];
        sr = fmaf(eyr, tr, fmaf(-eyi, ti, sr));
        si = fmaf(eyr, ti, fmaf(eyi, tr, si));
      }
      sr += __shfl_xor(sr, 1); sr += __shfl_xor(sr, 2);
      sr += __shfl_xor(sr, 4); sr += __shfl_xor(sr, 8);
      si += __shfl_xor(si, 1); si += __shfl_xor(si, 2);
      si += __shfl_xor(si, 4); si += __shfl_xor(si, 8);
      if (lo == 0) {
        float dc = DC[(size_t)kg * NTc + t] * (1.0f / 16384.0f); // ortho^2
        gSh[w * 16 + r + 8 * hi] = make_float2(sr * dc, si * dc);
      }
    }
    __syncthreads();

    // ---- u^T[y][k] = conj(Ey)[k,y] * g[k]  (B operand of adjoint GEMM) ----
#pragma unroll 4
    for (int i = 0; i < 64; ++i) {
      int e  = tid + i * 256;
      int y  = e >> 7;
      int kl = e & (KCH - 1);
      int kg = kg0 + kl;
      union { uint32_t u; _Float16 h[2]; } ey;
      ey.u = EyPt[(size_t)kg * NYc + y];
      float eyr = (float)ey.h[0], eyi = (float)ey.h[1];
      float2 g = gSh[kl];
      uTr[y * KCH + kl] = (_Float16)(eyr * g.x + eyi * g.y);
      uTi[y * KCH + kl] = (_Float16)(eyr * g.y - eyi * g.x);
    }
    __syncthreads();

    // ---- adjoint GEMM: xim[x,y] += conj(Ex)^T @ u  ------------------------
    // conj(Ex)^T: A_r = ExT_r, A_i = -ExT_i
    //   Dr += ExTr*ur + ExTi*ui ; Di += ExTr*ui - ExTi*ur
#pragma unroll
    for (int kk = 0; kk < KCH; kk += 32) {
      half16 Ar  = load_a_frag(ExTRt, NRADc, w * 16 + lo, kg0 + kk + hi * 8);
      half16 Ai  = load_a_frag(ExTIt, NRADc, w * 16 + lo, kg0 + kk + hi * 8);
      half16 AiN = hneg16(Ai);
#pragma unroll
      for (int n = 0; n < 8; ++n) {
        half16 Br = load_b_frag(uTr + (n * 16 + lo) * KCH + kk + hi * 8);
        half16 Bi = load_b_frag(uTi + (n * 16 + lo) * KCH + kk + hi * 8);
        xAr[n] = wmma_f16(Ar,  Br, xAr[n]);
        xAr[n] = wmma_f16(Ai,  Bi, xAr[n]);
        xAi[n] = wmma_f16(Ar,  Bi, xAi[n]);
        xAi[n] = wmma_f16(AiN, Br, xAi[n]);
      }
    }
  }

  // ---- coil combine: out += conj(smap) * xim  (atomic across coils) -------
#pragma unroll
  for (int r = 0; r < 8; ++r) {
    const int xg = w * 16 + r + 8 * hi;
#pragma unroll
    for (int n = 0; n < 8; ++n) {
      const int y = n * 16 + lo;
      float smr = CS[((size_t)c * 2 + 0) * NXc * NYc + (size_t)xg * NYc + y];
      float smi = CS[((size_t)c * 2 + 1) * NXc * NYc + (size_t)xg * NYc + y];
      float xr = xAr[n][r], xi = xAi[n][r];
      atomicAdd(out + (size_t)xg * NYc * NTc + (size_t)y * NTc + t,
                smr * xr + smi * xi);
      atomicAdd(out + nxyt + (size_t)xg * NYc * NTc + (size_t)y * NTc + t,
                smr * xi - smi * xr);
    }
  }
}

// ---------------------------------------------------------------------------
extern "C" void kernel_launch(void* const* d_in, const int* in_sizes, int n_in,
                              void* d_out, int out_size, void* d_ws, size_t ws_size,
                              hipStream_t stream) {
  (void)in_sizes; (void)n_in; (void)ws_size;
  const float* X  = (const float*)d_in[0];   // (2,NX,NY,NT)
  const float* KT = (const float*)d_in[1];   // (2,NRAD,NT)
  const float* CS = (const float*)d_in[2];   // (NC,2,NX,NY)
  const float* DC = (const float*)d_in[3];   // (NRAD,NT)
  float* out = (float*)d_out;

  // workspace layout (halves)
  const size_t S1 = (size_t)NTc * NRADc * NXc;           // 4,194,304
  _Float16* ExR  = (_Float16*)d_ws;
  _Float16* ExI  = ExR  + S1;
  _Float16* ExTR = ExI  + S1;
  _Float16* ExTI = ExTR + S1;
  uint32_t* EyP  = (uint32_t*)(ExTI + S1);               // S1 u32s

  // 1) phase tables
  {
    int total = (int)S1;
    build_tables<<<(total + 255) / 256, 256, 0, stream>>>(KT, ExR, ExI,
                                                          ExTR, ExTI, EyP);
  }
  // 2) zero output (coil combine is atomic-accumulate)
  zero_out<<<(out_size + 255) / 256, 256, 0, stream>>>(out, out_size);

  // 3) fused normal operator: one WG per (coil, frame)
  {
    dim3 grid(NCc, NTc);
    size_t shmem = (size_t)4 * NXc * NYc * sizeof(_Float16)   // srcT + uT
                 + (size_t)KCH * sizeof(float2);              // g
    adaga_kernel<<<grid, 256, shmem, stream>>>(X, CS, DC, ExR, ExI,
                                               ExTR, ExTI, EyP, out);
  }
}
